// StructureModule_80711025426901
// MI455X (gfx1250) — matrix operand order repaired and checked by reading, not compiled, not run on hardware
//
#include <hip/hip_runtime.h>
#include <math.h>

// ---------------------------------------------------------------------------
// Types for CDNA5 WMMA
// ---------------------------------------------------------------------------
typedef __attribute__((ext_vector_type(8)))  float  v8f;
typedef __attribute__((ext_vector_type(16))) __bf16 v16bf;

#define LSEQ 768
#define NCH  384
#define ECH  128
#define GRP  12
#define CATW 2112

// f32 pair -> packed bf16 (RTNE); native __bf16 conversion lets the backend
// use the hardware packed convert instead of manual bit-twiddling.
__device__ __forceinline__ unsigned int packbf(float a, float b) {
  union { __bf16 h[2]; unsigned int u; } r;
  r.h[0] = (__bf16)a;
  r.h[1] = (__bf16)b;
  return r.u;
}

// flags
#define GF_BIAS  1
#define GF_RELU  2
#define GF_ACC   4
#define GF_ARELU 8

// ---------------------------------------------------------------------------
// Generic bf16 WMMA GEMM:  C[M,N] = op( A[M,K] @ B[K,N] + bias + src )
// M multiple of 64, K multiple of 32, N arbitrary (bounds-checked).
// Block = 256 threads (8 waves), 64x64 tile; LDS tiles stored pre-packed as
// bf16 K-pairs (uint) so each WMMA fragment is 8 ds_load_b32.
// ---------------------------------------------------------------------------
__global__ __launch_bounds__(256) void gemm_bf16_k(
    const float* __restrict__ A, int lda,
    const float* __restrict__ B, int ldb,
    float* __restrict__ C, int ldc,
    const float* __restrict__ bias,
    const float* __restrict__ src, int ldsrc,
    int N, int K, int flags)
{
  __shared__ unsigned int sAp[64][16];   // [m][k2]  k2 = K-pair index
  __shared__ unsigned int sBp[16][64];   // [k2][n]
  const int tid  = threadIdx.x;
  const int lane = tid & 31, wave = tid >> 5;
  const int hi = lane >> 4, lo = lane & 15;
  const int bm = blockIdx.y * 64, bn = blockIdx.x * 64;
  const int mi = wave & 3, nj = wave >> 2;

  const v8f zero = {0.f,0.f,0.f,0.f,0.f,0.f,0.f,0.f};
  v8f acc0 = zero, acc1 = zero;

  const int ar = tid >> 2, ac = (tid & 3) * 8;      // A loader: row, 8 consecutive k
  const int bk2 = tid >> 4, bn0 = (tid & 15) * 4;   // B loader: k-pair, 4 n columns

  for (int kb = 0; kb < K; kb += 32) {
    // ---- A tile: 64 rows x 32 k, packed into 16 uints/row
    {
      const float* ap = A + (size_t)(bm + ar) * lda + kb + ac;
      float a[8];
#pragma unroll
      for (int i = 0; i < 8; i++) {
        float v = ap[i];
        if (flags & GF_ARELU) v = fmaxf(v, 0.f);
        a[i] = v;
      }
      uint4 uv;
      uv.x = packbf(a[0], a[1]); uv.y = packbf(a[2], a[3]);
      uv.z = packbf(a[4], a[5]); uv.w = packbf(a[6], a[7]);
      *(uint4*)&sAp[ar][(tid & 3) * 4] = uv;
    }
    // ---- B tile: 32 k x 64 n, packed along K pairs
    {
      const float* b0 = B + (size_t)(kb + 2 * bk2) * ldb + bn + bn0;
      const float* b1 = b0 + ldb;
      uint4 uv;
      unsigned int uu[4];
#pragma unroll
      for (int i = 0; i < 4; i++) {
        int n = bn + bn0 + i;
        float v0 = (n < N) ? b0[i] : 0.f;
        float v1 = (n < N) ? b1[i] : 0.f;
        uu[i] = packbf(v0, v1);
      }
      uv.x = uu[0]; uv.y = uu[1]; uv.z = uu[2]; uv.w = uu[3];
      *(uint4*)&sBp[bk2][bn0] = uv;
    }
    if (kb + 32 < K) __builtin_prefetch((const void*)(A + (size_t)(bm + ar) * lda + kb + 32), 0, 1);
    __syncthreads();

    union { unsigned int u[8]; v16bf v; } fa, fb0, fb1;
    // A fragment (16x32 bf16, ISA 7.12.2): lane(0-15)->K 0..7,16..23 ; lane(16-31)->K 8..15,24..31
    const unsigned int* arow = &sAp[mi * 16 + lo][0];
#pragma unroll
    for (int j = 0; j < 4; j++) { fa.u[j] = arow[hi * 4 + j]; fa.u[4 + j] = arow[8 + hi * 4 + j]; }
    // B fragment (32x16 bf16): lanes 0-15 K=0..7,16..23 ; lanes 16-31 K=8..15,24..31, col = lane&15
    const int nb = nj * 32;
#pragma unroll
    for (int j = 0; j < 8; j++) {
      int p = (j < 4) ? (hi * 4 + j) : (8 + hi * 4 + (j - 4));
      fb0.u[j] = sBp[p][nb + lo];
      fb1.u[j] = sBp[p][nb + 16 + lo];
    }
    acc0 = __builtin_amdgcn_wmma_f32_16x16x32_bf16(false, fa.v, false, fb0.v, (short)0, acc0, false, false);
    acc1 = __builtin_amdgcn_wmma_f32_16x16x32_bf16(false, fa.v, false, fb1.v, (short)0, acc1, false, false);
    __syncthreads();
  }

  v8f accs[2] = {acc0, acc1};
#pragma unroll
  for (int t2 = 0; t2 < 2; t2++) {
    int n = bn + nj * 32 + t2 * 16 + lo;
    if (n < N) {
      float bsv = (flags & GF_BIAS) ? bias[n] : 0.f;
#pragma unroll
      for (int r = 0; r < 8; r++) {
        int m = bm + mi * 16 + hi * 8 + r;  // D layout: VGPR r -> M = r + 8*(lane>>4)
        float v = accs[t2][r] + bsv;
        if (flags & GF_ACC)  v += src[(size_t)m * ldsrc + n];
        if (flags & GF_RELU) v = fmaxf(v, 0.f);
        C[(size_t)m * ldc + n] = v;
      }
    }
  }
}

// ---------------------------------------------------------------------------
// LayerNorm over last dim (block per row)
// ---------------------------------------------------------------------------
__global__ __launch_bounds__(256) void layernorm_k(
    const float* __restrict__ in, float* __restrict__ out,
    const float* __restrict__ g, const float* __restrict__ b, int dim)
{
  __shared__ float red[256];
  const int row = blockIdx.x, tid = threadIdx.x;
  const float* x = in + (size_t)row * dim;
  float s = 0.f, sq = 0.f;
  for (int c = tid; c < dim; c += 256) { float v = x[c]; s += v; sq += v * v; }
  red[tid] = s; __syncthreads();
  for (int o = 128; o > 0; o >>= 1) { if (tid < o) red[tid] += red[tid + o]; __syncthreads(); }
  float mean = red[0] / (float)dim; __syncthreads();
  red[tid] = sq; __syncthreads();
  for (int o = 128; o > 0; o >>= 1) { if (tid < o) red[tid] += red[tid + o]; __syncthreads(); }
  float var = red[0] / (float)dim - mean * mean;
  float r = rsqrtf(var + 1e-5f);
  float* o_ = out + (size_t)row * dim;
  for (int c = tid; c < dim; c += 256) o_[c] = (x[c] - mean) * r * g[c] + b[c];
}

// ---------------------------------------------------------------------------
// Edge preprocessing: per (l,m) row of Edge[.,.,128] compute LN stats (mu, rs)
// and bias[l,m,g] = LN(Edge[l,m,:]) @ wbias[:,g].  One wave per row.
// ---------------------------------------------------------------------------
__global__ __launch_bounds__(256) void edge_pre_k(
    const float* __restrict__ Edge, const float* __restrict__ lng,
    const float* __restrict__ lnb, const float* __restrict__ wb,
    float* __restrict__ mu, float* __restrict__ rs, float* __restrict__ biasb)
{
  const int row = blockIdx.x * 8 + (threadIdx.x >> 5);
  const int lane = threadIdx.x & 31;
  const float4 v = *(const float4*)(Edge + (size_t)row * ECH + lane * 4);
  float s  = v.x + v.y + v.z + v.w;
  float sq = v.x * v.x + v.y * v.y + v.z * v.z + v.w * v.w;
  for (int o = 16; o > 0; o >>= 1) { s += __shfl_xor(s, o); sq += __shfl_xor(sq, o); }
  const float mean = s * (1.f / 128.f);
  const float var  = sq * (1.f / 128.f) - mean * mean;
  const float r = rsqrtf(var + 1e-5f);
  if (lane == 0) { mu[row] = mean; rs[row] = r; }
  const int c0 = lane * 4;
  const float e0 = (v.x - mean) * r * lng[c0 + 0] + lnb[c0 + 0];
  const float e1 = (v.y - mean) * r * lng[c0 + 1] + lnb[c0 + 1];
  const float e2 = (v.z - mean) * r * lng[c0 + 2] + lnb[c0 + 2];
  const float e3 = (v.w - mean) * r * lng[c0 + 3] + lnb[c0 + 3];
  float bg[GRP];
#pragma unroll
  for (int g = 0; g < GRP; g++)
    bg[g] = e0 * wb[(c0 + 0) * GRP + g] + e1 * wb[(c0 + 1) * GRP + g]
          + e2 * wb[(c0 + 2) * GRP + g] + e3 * wb[(c0 + 3) * GRP + g];
#pragma unroll
  for (int g = 0; g < GRP; g++) {
    float t = bg[g];
    for (int o = 16; o > 0; o >>= 1) t += __shfl_xor(t, o);
    if (lane == 0) biasb[(size_t)row * GRP + g] = t;
  }
}

// ---------------------------------------------------------------------------
// Apply frames to local points: y = R x + t/10 for qv/kv/vv. One block per l.
// ---------------------------------------------------------------------------
__global__ __launch_bounds__(192) void apply_frames_k(
    const float* __restrict__ qkv, const float* __restrict__ Rb,
    const float* __restrict__ tb, float* __restrict__ qvg,
    float* __restrict__ kvg, float* __restrict__ vvg)
{
  const int l = blockIdx.x, tid = threadIdx.x;
  __shared__ float R[9], ts[3];
  if (tid < 9) R[tid] = Rb[l * 9 + tid];
  if (tid < 3) ts[tid] = tb[l * 3 + tid] * 0.1f;
  __syncthreads();
  const float* base = qkv + (size_t)l * 1152;
  const float* p; float* op;
  if (tid < 48)      { p = base + 576 + tid * 3;        op = qvg + (size_t)l * 144 + tid * 3; }
  else if (tid < 96) { p = base + 720 + (tid - 48) * 3; op = kvg + (size_t)l * 144 + (tid - 48) * 3; }
  else               { p = base + 864 + (tid - 96) * 3; op = vvg + (size_t)l * 288 + (tid - 96) * 3; }
  const float x0 = p[0], x1 = p[1], x2 = p[2];
  op[0] = R[0] * x0 + R[1] * x1 + R[2] * x2 + ts[0];
  op[1] = R[3] * x0 + R[4] * x1 + R[5] * x2 + ts[1];
  op[2] = R[6] * x0 + R[7] * x1 + R[8] * x2 + ts[2];
}

__global__ __launch_bounds__(32) void point_norms_k(
    const float* __restrict__ qvg, const float* __restrict__ kvg,
    float* __restrict__ qn, float* __restrict__ kn)
{
  const int l = blockIdx.x, tid = threadIdx.x;
  if (tid < GRP) {
    const float* p = qvg + (size_t)l * 144 + tid * 12;
    float s = 0.f;
#pragma unroll
    for (int i = 0; i < 12; i++) s += p[i] * p[i];
    qn[l * GRP + tid] = s;
  } else if (tid >= 16 && tid < 16 + GRP) {
    const int g = tid - 16;
    const float* p = kvg + (size_t)l * 144 + g * 12;
    float s = 0.f;
#pragma unroll
    for (int i = 0; i < 12; i++) s += p[i] * p[i];
    kn[l * GRP + g] = s;
  }
}

// ---------------------------------------------------------------------------
// Fused IPA attention. One block per query l (grid 768, 256 threads):
//   phase 1: logits[m][g]     (VALU; k/kv/bias are L2-resident)
//   phase 2: softmax over m per g, then pre-pack attn as bf16 K-pairs
//   phase 3: O_node, O_vec (frame-inverted), O_norm  (VALU dots)
//   phase 4: O_edge[g,c] = sum_m attn[m,g]*LN(Edge[l,m,c]) via bf16 WMMA,
//            Edge tile streamed HBM->LDS (LN fused) pre-packed as K-pairs.
// ---------------------------------------------------------------------------
__global__ __launch_bounds__(256) void ipa_attn_k(
    const float* __restrict__ Edge, const float* __restrict__ qkv,
    const float* __restrict__ qvg, const float* __restrict__ kvg,
    const float* __restrict__ vvg, const float* __restrict__ qn,
    const float* __restrict__ kn, const float* __restrict__ mu,
    const float* __restrict__ rs, const float* __restrict__ biasb,
    const float* __restrict__ lng, const float* __restrict__ lnb,
    const float* __restrict__ gamma, const float* __restrict__ Rb,
    const float* __restrict__ tb, float* __restrict__ catb)
{
  const int l = blockIdx.x;
  const int tid = threadIdx.x;
  const int lane = tid & 31, wave = tid >> 5;
  const int hi = lane >> 4, lo = lane & 15;

  __shared__ float s_attn[LSEQ][GRP];          // logits -> probs (f32)
  __shared__ unsigned int s_ap[LSEQ / 2][16];  // probs packed bf16 K-pairs, g padded to 16
  __shared__ unsigned int s_ep[16][ECH];       // LN(Edge) tile packed bf16 K-pairs
  __shared__ float s_q[GRP * 16];
  __shared__ float s_qv[GRP * 12];
  __shared__ float s_qn[GRP];
  __shared__ float s_gam[GRP];
  __shared__ float s_R[9], s_t[3];
  __shared__ float s_g[ECH], s_b[ECH];
  __shared__ float red[256];

  // stage per-l data
  if (tid < 192) s_q[tid] = qkv[(size_t)l * 1152 + tid];
  if (tid < 144) s_qv[tid] = qvg[(size_t)l * 144 + tid];
  if (tid < GRP) {
    s_qn[tid] = qn[l * GRP + tid];
    float x = gamma[tid];
    s_gam[tid] = (x > 20.f) ? x : log1pf(expf(x));  // softplus
  }
  if (tid < 9) s_R[tid] = Rb[l * 9 + tid];
  if (tid < 3) s_t[tid] = tb[l * 3 + tid] * 0.1f;
  if (tid < ECH) { s_g[tid] = lng[tid]; s_b[tid] = lnb[tid]; }
  __syncthreads();

  const float WC = 0.2357022603955158f;   // sqrt(2/(9*QP)), QP=4
  const float WL = 0.5773502691896258f;   // sqrt(1/3)

  // phase 1: logits
  for (int idx = tid; idx < LSEQ * GRP; idx += 256) {
    const int m = idx / GRP, g = idx - m * GRP;
    const float* kp = qkv + (size_t)m * 1152 + 192 + g * 16;
    float qk = 0.f;
#pragma unroll
    for (int c = 0; c < 16; c++) qk += s_q[g * 16 + c] * kp[c];
    const float* kvp = kvg + (size_t)m * 144 + g * 12;
    const float* qvp = s_qv + g * 12;
    float qkp = 0.f;
#pragma unroll
    for (int c = 0; c < 12; c++) qkp += qvp[c] * kvp[c];
    const float dist = s_qn[g] + kn[m * GRP + g] - 2.f * qkp;
    const float a = qk * 0.25f + biasb[((size_t)l * LSEQ + m) * GRP + g]
                  - s_gam[g] * (WC * 0.5f) * dist;
    s_attn[m][g] = WL * a;
  }
  __syncthreads();

  // phase 2: softmax over m, per group
  for (int g = 0; g < GRP; g++) {
    float mx = -1e30f;
    for (int m = tid; m < LSEQ; m += 256) mx = fmaxf(mx, s_attn[m][g]);
    red[tid] = mx; __syncthreads();
    for (int o = 128; o > 0; o >>= 1) { if (tid < o) red[tid] = fmaxf(red[tid], red[tid + o]); __syncthreads(); }
    mx = red[0]; __syncthreads();
    float sm = 0.f;
    for (int m = tid; m < LSEQ; m += 256) { float e = expf(s_attn[m][g] - mx); s_attn[m][g] = e; sm += e; }
    red[tid] = sm; __syncthreads();
    for (int o = 128; o > 0; o >>= 1) { if (tid < o) red[tid] += red[tid + o]; __syncthreads(); }
    const float inv = 1.f / red[0]; __syncthreads();
    for (int m = tid; m < LSEQ; m += 256) s_attn[m][g] *= inv;
    __syncthreads();
  }

  // pre-pack attention weights for the WMMA A-matrix (M=g padded to 16, K=m)
  for (int idx = tid; idx < (LSEQ / 2) * 16; idx += 256) {
    const int k2 = idx >> 4, g = idx & 15;
    s_ap[k2][g] = (g < GRP) ? packbf(s_attn[2 * k2][g], s_attn[2 * k2 + 1][g]) : 0u;
  }

  float* cb = catb + (size_t)l * CATW;

  // phase 3a: O_node (12 groups x 16 ch)
  for (int idx = tid; idx < 192; idx += 256) {
    const int g = idx >> 4, c = idx & 15;
    float s = 0.f;
    for (int m = 0; m < LSEQ; m++) s += s_attn[m][g] * qkv[(size_t)m * 1152 + 384 + g * 16 + c];
    cb[1536 + idx] = s;
  }
  // phase 3b: O_vec (frame-inverted) + O_norm (12 x 8 points)
  for (int idx = tid; idx < 96; idx += 256) {
    const int g = idx >> 3, p = idx & 7;
    float s0 = 0.f, s1 = 0.f, s2 = 0.f;
    for (int m = 0; m < LSEQ; m++) {
      const float a = s_attn[m][g];
      const float* q = vvg + (size_t)m * 288 + g * 24 + p * 3;
      s0 += a * q[0]; s1 += a * q[1]; s2 += a * q[2];
    }
    s0 -= s_t[0]; s1 -= s_t[1]; s2 -= s_t[2];
    const float o0 = s_R[0] * s0 + s_R[3] * s1 + s_R[6] * s2;  // R^T
    const float o1 = s_R[1] * s0 + s_R[4] * s1 + s_R[7] * s2;
    const float o2 = s_R[2] * s0 + s_R[5] * s1 + s_R[8] * s2;
    cb[1728 + g * 24 + p * 3 + 0] = o0;
    cb[1728 + g * 24 + p * 3 + 1] = o1;
    cb[1728 + g * 24 + p * 3 + 2] = o2;
    cb[2016 + g * 8 + p] = sqrtf(o0 * o0 + o1 * o1 + o2 * o2 + 1e-8f);
  }

  // phase 4: O_edge via WMMA
  const v8f zero = {0.f,0.f,0.f,0.f,0.f,0.f,0.f,0.f};
  v8f acc = zero;
  for (int kb = 0; kb < LSEQ; kb += 32) {
    __syncthreads();
    // stream Edge tile [32 x 128] HBM -> LDS with LN fused, packed as bf16 K-pairs
    for (int task = tid; task < 512; task += 256) {
      const int p  = task >> 5;          // K-pair 0..15
      const int c4 = (task & 31) * 4;    // column base
      const int m0 = kb + 2 * p;
      const float4 e0 = *(const float4*)(Edge + ((size_t)l * LSEQ + m0) * ECH + c4);
      const float4 e1 = *(const float4*)(Edge + ((size_t)l * LSEQ + m0 + 1) * ECH + c4);
      const float mu0 = mu[(size_t)l * LSEQ + m0], rs0 = rs[(size_t)l * LSEQ + m0];
      const float mu1 = mu[(size_t)l * LSEQ + m0 + 1], rs1 = rs[(size_t)l * LSEQ + m0 + 1];
      uint4 uv;
      uv.x = packbf((e0.x - mu0) * rs0 * s_g[c4 + 0] + s_b[c4 + 0],
                    (e1.x - mu1) * rs1 * s_g[c4 + 0] + s_b[c4 + 0]);
      uv.y = packbf((e0.y - mu0) * rs0 * s_g[c4 + 1] + s_b[c4 + 1],
                    (e1.y - mu1) * rs1 * s_g[c4 + 1] + s_b[c4 + 1]);
      uv.z = packbf((e0.z - mu0) * rs0 * s_g[c4 + 2] + s_b[c4 + 2],
                    (e1.z - mu1) * rs1 * s_g[c4 + 2] + s_b[c4 + 2]);
      uv.w = packbf((e0.w - mu0) * rs0 * s_g[c4 + 3] + s_b[c4 + 3],
                    (e1.w - mu1) * rs1 * s_g[c4 + 3] + s_b[c4 + 3]);
      *(uint4*)&s_ep[p][c4] = uv;
    }
    if (kb + 32 < LSEQ)
      __builtin_prefetch((const void*)(Edge + ((size_t)l * LSEQ + kb + 32 + (tid >> 3)) * ECH + (tid & 7) * 16), 0, 1);
    __syncthreads();

    union { unsigned int u[8]; v16bf v; } fa, fb;
    const int k2b = kb >> 1;
#pragma unroll
    for (int j = 0; j < 8; j++) {
      const int pp = (j < 4) ? (hi * 4 + j) : (8 + hi * 4 + (j - 4));
      fa.u[j] = s_ap[k2b + pp][lo];
    }
    const int n = wave * 16 + lo;
#pragma unroll
    for (int j = 0; j < 8; j++) fb.u[j] = s_ep[(j < 4) ? (hi * 4 + j) : (8 + hi * 4 + (j - 4))][n];
    acc = __builtin_amdgcn_wmma_f32_16x16x32_bf16(false, fa.v, false, fb.v, (short)0, acc, false, false);
  }
  {
    const int n = wave * 16 + lo;
#pragma unroll
    for (int r = 0; r < 8; r++) {
      const int g = hi * 8 + r;
      if (g < GRP) cb[g * ECH + n] = acc[r];
    }
  }
}

// ---------------------------------------------------------------------------
// Backbone update: u = Node @ s2t_w + s2t_b, quaternion -> Ru, tu; compose.
// ---------------------------------------------------------------------------
__global__ __launch_bounds__(192) void backbone_k(
    const float* __restrict__ node, const float* __restrict__ w,
    const float* __restrict__ b, float* __restrict__ Rb, float* __restrict__ tb)
{
  const int l = blockIdx.x;
  const int wv = threadIdx.x >> 5, lane = threadIdx.x & 31;
  __shared__ float u[6];
  const float* n = node + (size_t)l * NCH;
  float s = 0.f;
  for (int c = lane; c < NCH; c += 32) s += n[c] * w[c * 6 + wv];
  for (int o = 16; o > 0; o >>= 1) s += __shfl_xor(s, o);
  if (lane == 0) u[wv] = s + b[wv];
  __syncthreads();
  if (threadIdx.x == 0) {
    const float bq = u[0], c = u[1], d = u[2];
    const float inv = 1.f / (1.f + bq * bq + c * c + d * d);
    float Ru[9];
    Ru[0] = (1.f + bq * bq - c * c - d * d) * inv; Ru[1] = (2.f * bq * c - 2.f * d) * inv; Ru[2] = (2.f * bq * d + 2.f * c) * inv;
    Ru[3] = (2.f * bq * c + 2.f * d) * inv; Ru[4] = (1.f - bq * bq + c * c - d * d) * inv; Ru[5] = (2.f * c * d - 2.f * bq) * inv;
    Ru[6] = (2.f * bq * d - 2.f * c) * inv; Ru[7] = (2.f * c * d + 2.f * bq) * inv; Ru[8] = (1.f - bq * bq - c * c + d * d) * inv;
    const float tu0 = 10.f * u[3], tu1 = 10.f * u[4], tu2 = 10.f * u[5];
    float* R = Rb + l * 9; float* t = tb + l * 3;
    float r[9];
#pragma unroll
    for (int i = 0; i < 9; i++) r[i] = R[i];
    t[0] += r[0] * tu0 + r[1] * tu1 + r[2] * tu2;
    t[1] += r[3] * tu0 + r[4] * tu1 + r[5] * tu2;
    t[2] += r[6] * tu0 + r[7] * tu1 + r[8] * tu2;
#pragma unroll
    for (int i = 0; i < 3; i++)
#pragma unroll
      for (int j = 0; j < 3; j++)
        R[i * 3 + j] = r[i * 3 + 0] * Ru[0 * 3 + j] + r[i * 3 + 1] * Ru[1 * 3 + j] + r[i * 3 + 2] * Ru[2 * 3 + j];
  }
}

__global__ void init_frames_k(float* __restrict__ Rb, float* __restrict__ tb) {
  const int l = blockIdx.x * blockDim.x + threadIdx.x;
  if (l < LSEQ) {
    float* R = Rb + l * 9;
    R[0] = 1.f; R[1] = 0.f; R[2] = 0.f;
    R[3] = 0.f; R[4] = 1.f; R[5] = 0.f;
    R[6] = 0.f; R[7] = 0.f; R[8] = 1.f;
    tb[l * 3 + 0] = 0.f; tb[l * 3 + 1] = 0.f; tb[l * 3 + 2] = 0.f;
  }
}

__global__ __launch_bounds__(96) void finalize_k(
    const float* __restrict__ pl, const float* __restrict__ tors,
    const float* __restrict__ tb, float* __restrict__ out)
{
  const int l = blockIdx.x, i = threadIdx.x;
  if (i < 67) {
    float v = (i < 50) ? pl[l * 50 + i] : (i < 64) ? tors[l * 14 + (i - 50)] : tb[l * 3 + (i - 64)];
    out[(size_t)l * 67 + i] = v;
  }
}

// ---------------------------------------------------------------------------
// Host orchestration
// ---------------------------------------------------------------------------
extern "C" void kernel_launch(void* const* d_in, const int* in_sizes, int n_in,
                              void* d_out, int out_size, void* d_ws, size_t ws_size,
                              hipStream_t stream) {
  (void)in_sizes; (void)n_in; (void)out_size; (void)ws_size;
  const float* Node_in  = (const float*)d_in[0];
  const float* Edge     = (const float*)d_in[1];
  const float* ln_s_g   = (const float*)d_in[2];
  const float* ln_s_b   = (const float*)d_in[3];
  const float* ln_e_g   = (const float*)d_in[4];
  const float* ln_e_b   = (const float*)d_in[5];
  const float* proj_s_w = (const float*)d_in[6];
  const float* proj_s_b = (const float*)d_in[7];
  const float* wq   = (const float*)d_in[8];
  const float* wk   = (const float*)d_in[9];
  const float* wv   = (const float*)d_in[10];
  const float* wqv  = (const float*)d_in[11];
  const float* wkv  = (const float*)d_in[12];
  const float* wvv  = (const float*)d_in[13];
  const float* wbias= (const float*)d_in[14];
  const float* gamma= (const float*)d_in[15];
  const float* woutp= (const float*)d_in[16];
  const float* bout = (const float*)d_in[17];
  const float* tr_ln_g = (const float*)d_in[18];
  const float* tr_ln_b = (const float*)d_in[19];
  const float* tr_w1 = (const float*)d_in[20];
  const float* tr_b1 = (const float*)d_in[21];
  const float* tr_w2 = (const float*)d_in[22];
  const float* tr_b2 = (const float*)d_in[23];
  const float* tr_w3 = (const float*)d_in[24];
  const float* tr_b3 = (const float*)d_in[25];
  const float* ln_itr_g = (const float*)d_in[26];
  const float* ln_itr_b = (const float*)d_in[27];
  const float* s2t_w = (const float*)d_in[28];
  const float* s2t_b = (const float*)d_in[29];
  const float* t_p0w = (const float*)d_in[30];
  const float* t_p0b = (const float*)d_in[31];
  const float* t_p1w = (const float*)d_in[32];
  const float* t_p1b = (const float*)d_in[33];
  const float* t_a1w = (const float*)d_in[34];
  const float* t_a1b = (const float*)d_in[35];
  const float* t_a2w = (const float*)d_in[36];
  const float* t_a2b = (const float*)d_in[37];
  const float* t_b1w = (const float*)d_in[38];
  const float* t_b1b = (const float*)d_in[39];
  const float* t_b2w = (const float*)d_in[40];
  const float* t_b2b = (const float*)d_in[41];
  const float* t_ow  = (const float*)d_in[42];
  const float* t_ob  = (const float*)d_in[43];
  const float* pl_ln_g = (const float*)d_in[44];
  const float* pl_ln_b = (const float*)d_in[45];
  const float* pl_w1 = (const float*)d_in[46];
  const float* pl_b1 = (const float*)d_in[47];
  const float* pl_w2 = (const float*)d_in[48];
  const float* pl_b2 = (const float*)d_in[49];
  const float* pl_w3 = (const float*)d_in[50];
  const float* pl_b3 = (const float*)d_in[51];
  float* out = (float*)d_out;

  // workspace carve-up
  float* ws = (float*)d_ws;
  size_t off = 0;
  auto alloc = [&](size_t nelem) { float* p = ws + off; off += (nelem + 3) & ~(size_t)3; return p; };
  float* node_raw = alloc((size_t)LSEQ * NCH);
  float* node     = alloc((size_t)LSEQ * NCH);
  float* xbuf     = alloc((size_t)LSEQ * NCH);
  float* hA       = alloc((size_t)LSEQ * NCH);
  float* hB       = alloc((size_t)LSEQ * NCH);
  float* qkv      = alloc((size_t)LSEQ * 1152);
  float* qvg      = alloc((size_t)LSEQ * 144);
  float* kvg      = alloc((size_t)LSEQ * 144);
  float* vvg      = alloc((size_t)LSEQ * 288);
  float* qn       = alloc((size_t)LSEQ * GRP);
  float* kn       = alloc((size_t)LSEQ * GRP);
  float* mub      = alloc((size_t)LSEQ * LSEQ);
  float* rsb      = alloc((size_t)LSEQ * LSEQ);
  float* biasb    = alloc((size_t)LSEQ * LSEQ * GRP);
  float* catb     = alloc((size_t)LSEQ * CATW);
  float* Rbuf     = alloc((size_t)LSEQ * 9);
  float* tbuf     = alloc((size_t)LSEQ * 3);
  float* x0       = alloc((size_t)LSEQ * 128);
  float* x1       = alloc((size_t)LSEQ * 128);
  float* hT       = alloc((size_t)LSEQ * 128);
  float* tors     = alloc((size_t)LSEQ * 14);
  float* plb      = alloc((size_t)LSEQ * 50);

  auto gemm = [&](const float* A, int lda, const float* B, int ldb, float* C, int ldc,
                  const float* bias, const float* src, int ldsrc, int N, int K, int flags) {
    dim3 grid((N + 63) / 64, LSEQ / 64);
    gemm_bf16_k<<<grid, dim3(256), 0, stream>>>(A, lda, B, ldb, C, ldc, bias, src, ldsrc, N, K, flags);
  };
  auto lnorm = [&](const float* in, float* o, const float* g, const float* b) {
    layernorm_k<<<dim3(LSEQ), dim3(256), 0, stream>>>(in, o, g, b, NCH);
  };

  // ---- pre-loop -----------------------------------------------------------
  lnorm(Node_in, node_raw, ln_s_g, ln_s_b);
  gemm(node_raw, NCH, proj_s_w, NCH, node, NCH, proj_s_b, nullptr, 0, NCH, NCH, GF_BIAS);
  init_frames_k<<<dim3(3), dim3(256), 0, stream>>>(Rbuf, tbuf);
  edge_pre_k<<<dim3((LSEQ * LSEQ) / 8), dim3(256), 0, stream>>>(Edge, ln_e_g, ln_e_b, wbias, mub, rsb, biasb);

  // ---- 8 IPA blocks -------------------------------------------------------
  for (int blk = 0; blk < 8; blk++) {
    gemm(node, NCH, wq,  192, qkv + 0,   1152, nullptr, nullptr, 0, 192, NCH, 0);
    gemm(node, NCH, wk,  192, qkv + 192, 1152, nullptr, nullptr, 0, 192, NCH, 0);
    gemm(node, NCH, wv,  192, qkv + 384, 1152, nullptr, nullptr, 0, 192, NCH, 0);
    gemm(node, NCH, wqv, 144, qkv + 576, 1152, nullptr, nullptr, 0, 144, NCH, 0);
    gemm(node, NCH, wkv, 144, qkv + 720, 1152, nullptr, nullptr, 0, 144, NCH, 0);
    gemm(node, NCH, wvv, 288, qkv + 864, 1152, nullptr, nullptr, 0, 288, NCH, 0);
    apply_frames_k<<<dim3(LSEQ), dim3(192), 0, stream>>>(qkv, Rbuf, tbuf, qvg, kvg, vvg);
    point_norms_k<<<dim3(LSEQ), dim3(32), 0, stream>>>(qvg, kvg, qn, kn);
    ipa_attn_k<<<dim3(LSEQ), dim3(256), 0, stream>>>(Edge, qkv, qvg, kvg, vvg, qn, kn,
                                                     mub, rsb, biasb, ln_e_g, ln_e_b,
                                                     gamma, Rbuf, tbuf, catb);
    gemm(catb, CATW, woutp, NCH, node, NCH, bout, node, NCH, NCH, CATW, GF_BIAS | GF_ACC);
    lnorm(node, node, ln_itr_g, ln_itr_b);
    lnorm(node, xbuf, tr_ln_g, tr_ln_b);
    gemm(xbuf, NCH, tr_w1, NCH, hA, NCH, tr_b1, nullptr, 0, NCH, NCH, GF_BIAS | GF_RELU);
    gemm(hA,   NCH, tr_w2, NCH, hB, NCH, tr_b2, nullptr, 0, NCH, NCH, GF_BIAS | GF_RELU);
    gemm(hB,   NCH, tr_w3, NCH, node, NCH, tr_b3, xbuf, NCH, NCH, NCH, GF_BIAS | GF_ACC);
    backbone_k<<<dim3(LSEQ), dim3(192), 0, stream>>>(node, s2t_w, s2t_b, Rbuf, tbuf);
  }

  // ---- torsion head -------------------------------------------------------
  gemm(node,     NCH, t_p0w, 128, x0, 128, t_p0b, nullptr, 0, 128, NCH, GF_BIAS);
  gemm(node_raw, NCH, t_p1w, 128, x0, 128, t_p1b, x0, 128, 128, NCH, GF_BIAS | GF_ACC);
  gemm(x0, 128, t_a1w, 128, hT, 128, t_a1b, nullptr, 0, 128, 128, GF_BIAS | GF_RELU | GF_ARELU);
  gemm(hT, 128, t_a2w, 128, x1, 128, t_a2b, x0, 128, 128, 128, GF_BIAS | GF_ACC);
  gemm(x1, 128, t_b1w, 128, hT, 128, t_b1b, nullptr, 0, 128, 128, GF_BIAS | GF_RELU | GF_ARELU);
  gemm(hT, 128, t_b2w, 128, x0, 128, t_b2b, x1, 128, 128, 128, GF_BIAS | GF_ACC);
  gemm(x0, 128, t_ow, 14, tors, 14, t_ob, nullptr, 0, 14, 128, GF_BIAS | GF_ARELU);

  // ---- pLDDT head ---------------------------------------------------------
  lnorm(node, xbuf, pl_ln_g, pl_ln_b);
  gemm(xbuf, NCH, pl_w1, 128, hT, 128, pl_b1, nullptr, 0, 128, NCH, GF_BIAS | GF_RELU);
  gemm(hT, 128, pl_w2, 128, x1, 128, pl_b2, nullptr, 0, 128, 128, GF_BIAS | GF_RELU);
  gemm(x1, 128, pl_w3, 50, plb, 50, pl_b3, nullptr, 0, 50, 128, GF_BIAS);

  finalize_k<<<dim3(LSEQ), dim3(96), 0, stream>>>(plb, tors, tbuf, out);
}
